// AggregateEdgesFromNodes_188978561162
// MI455X (gfx1250) — compile-verified, matrix operand order checked by reading, not run on hardware
//
#include <hip/hip_runtime.h>

// relu([src_feat | dst_feat | dist_feat] @ W + b)  -- M=800000, K=384, N=128
// Split-bf16 (hi+lo) WMMA GEMM: ah*bh + ah*bl + al*bh accumulated in f32.

#define HIDDEN 128
#define KTOT   384
#define LDA    392   // padded K-stride (bf16 elems): 392*2B = 784B = 196 dwords, 196%64=4 -> conflict-free fragment reads

// LDS layout (byte offsets), all 16B aligned
#define WT_HI_OFF 0
#define WT_LO_OFF (WT_HI_OFF + HIDDEN * LDA * 2)   // 100352
#define A_HI_OFF  (WT_LO_OFF + HIDDEN * LDA * 2)   // 200704
#define A_LO_OFF  (A_HI_OFF  + 16 * LDA * 2)       // 213248
#define BIAS_OFF  (A_LO_OFF  + 16 * LDA * 2)       // 225792
#define SMEM_BYTES (BIAS_OFF + HIDDEN * 4)         // 226304 < 320KB/WGP

typedef __attribute__((ext_vector_type(16))) __bf16        v16bf;
typedef __attribute__((ext_vector_type(8)))  float         v8f;
typedef __attribute__((ext_vector_type(4)))  float         f32x4;
typedef __attribute__((ext_vector_type(4)))  unsigned int  u32x4;
typedef __attribute__((ext_vector_type(2)))  unsigned int  u32x2;

__device__ __forceinline__ unsigned int f2bf_rne(float f) {
  unsigned int u = __float_as_uint(f);
  return (u + 0x7FFFu + ((u >> 16) & 1u)) >> 16;
}
// x = bf16(hi) + bf16(lo) + O(2^-16 * |x|)
__device__ __forceinline__ void split2(float x, unsigned int& h, unsigned int& l) {
  h = f2bf_rne(x);
  float hf = __uint_as_float(h << 16);
  l = f2bf_rne(x - hf);
}

// Load one 16x16x32 A/B fragment (16 bf16 per lane) from LDS.
// Elements 0..7  -> VGPR0..3 : K = kb .. kb+7   (contiguous)
// Elements 8..15 -> VGPR4..7 : K = kb+16 .. kb+23
// (kb = 0 for lanes 0..15, kb = 8 for lanes 16..31 -- handled by caller pointer)
__device__ __forceinline__ v16bf load_frag(const unsigned short* p) {
  union { u32x4 u[2]; v16bf v; } t;
  t.u[0] = *(const u32x4*)(p);
  t.u[1] = *(const u32x4*)(p + 16);
  return t.v;
}

__global__ __launch_bounds__(256, 1)
void edge_mlp_wmma(const float* __restrict__ nef,     // [850000,128]
                   const float* __restrict__ dist,    // [800000,128]
                   const int*   __restrict__ srcs,    // [800000]
                   const int*   __restrict__ dsts,    // [800000]
                   const float* __restrict__ W,       // [384,128] row-major
                   const float* __restrict__ bias,    // [128]
                   float*       __restrict__ out,     // [800000,128]
                   int n_edges)
{
  extern __shared__ char smem[];
  unsigned short* wt_hi = (unsigned short*)(smem + WT_HI_OFF);
  unsigned short* wt_lo = (unsigned short*)(smem + WT_LO_OFF);
  unsigned short* a_hi  = (unsigned short*)(smem + A_HI_OFF);
  unsigned short* a_lo  = (unsigned short*)(smem + A_LO_OFF);
  float*          bsh   = (float*)(smem + BIAS_OFF);

  const int tid  = threadIdx.x;
  const int wave = tid >> 5;      // 0..7, each owns a 16-col output slice
  const int lane = tid & 31;

  // ---- one-time per block: W[k][n] -> transposed split-bf16 planes WT[n][k]
  for (int idx = tid; idx < KTOT * HIDDEN; idx += 256) {
    int k = idx >> 7;
    int n = idx & (HIDDEN - 1);
    unsigned int h, l;
    split2(W[idx], h, l);
    wt_hi[n * LDA + k] = (unsigned short)h;
    wt_lo[n * LDA + k] = (unsigned short)l;
  }
  if (tid < HIDDEN) bsh[tid] = bias[tid];

  const int num_tiles = (n_edges + 15) >> 4;

  // per-wave WMMA fragment pointers
  const int c    = lane & 15;              // A row / B col within 16x16 tile
  const int ksel = (lane >> 4) << 3;       // lanes 16..31: K base +8
  const unsigned short* arow_h = a_hi + c * LDA + ksel;
  const unsigned short* arow_l = a_lo + c * LDA + ksel;
  const int ncol = (wave << 4) + c;        // global output column
  const unsigned short* brow_h = wt_hi + ncol * LDA + ksel;
  const unsigned short* brow_l = wt_lo + ncol * LDA + ksel;

  f32x4 pf[6];  // prefetched gather data: 6 rows/wave, 4 floats/lane

  // gather one tile: 48 rows (16 edges x {src,dst,dist}); wave w loads rows w, w+8, ... w+40
  auto prefetch = [&](int tt) {
    long long be = (long long)tt << 4;
#pragma unroll
    for (int j = 0; j < 6; ++j) {
      int r = wave + (j << 3);             // 0..47
      int e = r & 15;
      int which = r >> 4;                  // 0=src, 1=dst, 2=dist
      long long edge = be + e;
      if (edge > (long long)n_edges - 1) edge = (long long)n_edges - 1;
      const float* p;
      if (which == 0)      p = nef  + (long long)srcs[edge] * HIDDEN;
      else if (which == 1) p = nef  + (long long)dsts[edge] * HIDDEN;
      else                 p = dist + edge * HIDDEN;
      pf[j] = ((const f32x4*)p)[lane];     // global_load_b128
    }
  };

  int t = blockIdx.x;
  if (t < num_tiles) prefetch(t);

  for (; t < num_tiles; t += gridDim.x) {
    __syncthreads();   // previous tile's WMMA phase done reading A planes

    // convert prefetched rows -> split-bf16 A planes (K order: src|dst|dist matches W rows)
#pragma unroll
    for (int j = 0; j < 6; ++j) {
      int r = wave + (j << 3);
      int e = r & 15;
      int k0 = ((r >> 4) << 7) + (lane << 2);   // which*128 + lane*4
      f32x4 v = pf[j];
      unsigned int h0,h1,h2,h3,l0,l1,l2,l3;
      split2(v.x, h0, l0); split2(v.y, h1, l1);
      split2(v.z, h2, l2); split2(v.w, h3, l3);
      u32x2 hp, lp;
      hp.x = h0 | (h1 << 16); hp.y = h2 | (h3 << 16);
      lp.x = l0 | (l1 << 16); lp.y = l2 | (l3 << 16);
      *(u32x2*)(a_hi + e * LDA + k0) = hp;      // ds_store_b64
      *(u32x2*)(a_lo + e * LDA + k0) = lp;
    }
    __syncthreads();   // A planes ready

    int tn = t + (int)gridDim.x;
    if (tn < num_tiles) prefetch(tn);  // overlap next gather with WMMA phase

    v8f acc = {};
#pragma unroll
    for (int kk = 0; kk < KTOT; kk += 32) {
      v16bf ah = load_frag(arow_h + kk);
      v16bf al = load_frag(arow_l + kk);
      v16bf bh = load_frag(brow_h + kk);
      v16bf bl = load_frag(brow_l + kk);
      // D = A*B + C, f32 accumulation; three split terms (al*bl ~2^-16, dropped)
      acc = __builtin_amdgcn_wmma_f32_16x16x32_bf16(false, ah, false, bh, (short)0, acc, false, false);
      acc = __builtin_amdgcn_wmma_f32_16x16x32_bf16(false, ah, false, bl, (short)0, acc, false, false);
      acc = __builtin_amdgcn_wmma_f32_16x16x32_bf16(false, al, false, bh, (short)0, acc, false, false);
    }

    // epilogue: C/D layout: VGPR j, lanes 0-15 -> M=j, lanes 16-31 -> M=8+j; N = lane&15
    float bn = bsh[ncol];
    long long be = (long long)t << 4;
    int mbase = (lane >> 4) << 3;
#pragma unroll
    for (int j = 0; j < 8; ++j) {
      long long m = be + mbase + j;
      if (m < n_edges) {
        float v = acc[j] + bn;
        out[m * HIDDEN + ncol] = v > 0.f ? v : 0.f;
      }
    }
  }
}

extern "C" void kernel_launch(void* const* d_in, const int* in_sizes, int n_in,
                              void* d_out, int out_size, void* d_ws, size_t ws_size,
                              hipStream_t stream) {
  const float* nef  = (const float*)d_in[0];
  const float* dist = (const float*)d_in[1];
  const int*   srcs = (const int*)d_in[2];   // jnp int (x64 disabled -> int32)
  const int*   dsts = (const int*)d_in[3];
  const float* W    = (const float*)d_in[4];
  const float* b    = (const float*)d_in[5];
  float*       out  = (float*)d_out;

  int n_edges = in_sizes[2];
  if (n_edges <= 0) return;
  int num_tiles = (n_edges + 15) / 16;
  int blocks = num_tiles < 1024 ? num_tiles : 1024;

  // allow >default dynamic LDS (host-side attribute set; not a stream op, capture-safe)
  hipFuncSetAttribute((const void*)edge_mlp_wmma,
                      hipFuncAttributeMaxDynamicSharedMemorySize, SMEM_BYTES);

  edge_mlp_wmma<<<dim3(blocks), dim3(256), SMEM_BYTES, stream>>>(
      nef, dist, srcs, dsts, W, b, out, n_edges);
}
